// VanillaRNN_28363964023515
// MI455X (gfx1250) — matrix operand order repaired and checked by reading, not compiled, not run on hardware
//
#include <hip/hip_runtime.h>
#include <hip/hip_bf16.h>

// VanillaRNN on gfx1250:  h_{t+1} = tanh(x_t (x) Wx + h_t @ Wh + bh),  out = h_S @ Wo + bo
// B=1024, S=1024, H=512, C=10.
//
// Each workgroup owns a 16-row batch tile and runs the full 1024-step
// recurrence locally (h state in LDS bf16, double buffered; two s_barriers
// per step). Per step, 8 waves each compute a 16x64 slice of h@Wh with
// v_wmma_f32_16x16x32_bf16 (4 N-tiles x 16 K-chunks).
//
// Wh is pre-swizzled once (fp32->bf16) into WMMA B-fragments with contiguous k (16*(lane/16) + i), the order the A loads use.
// K-chunks 0..7 (256KB) are parked in LDS at kernel start (LDS budget:
// 33KB h + 256KB B = 289KB of the WGP's 320KB); K-chunks 8..15 stream from
// the L2-resident fragment buffer every step (16GB total over the run).
// Offsets are laundered through empty asm each iteration so the compiler
// cannot hoist the loop-invariant weight loads into VGPRs and spill
// (which it did last round: scratch_load_b128 reloads).

typedef __attribute__((ext_vector_type(16))) __bf16 v16bf;
typedef __attribute__((ext_vector_type(8)))  __bf16 v8bf;
typedef __attribute__((ext_vector_type(8)))  float  v8f;

#define SEQ     1024
#define HID     512
#define NCLS    10
#define BTILE   16
#define HSTRIDE 520                         // 512 + 8 halves pad: 16B aligned, no LDS bank conflicts

#define HBUF_BYTES (2 * BTILE * HSTRIDE * 2)   // 33280
#define XS_OFF     HBUF_BYTES                  // 33280 (16B aligned)
#define BSH_OFF    (XS_OFF + 64)               // 33344 (16B aligned)
#define BSH_BYTES  (8 * 8 * 4 * 32 * 16 * 2)   // 262144 (8 waves x kc0..7 x 4 nt x 32 lanes x 16 bf16)
#define SMEM_BYTES (BSH_OFF + BSH_BYTES)       // 295488

__device__ __forceinline__ float fast_tanh(float v) {
#if __has_builtin(__builtin_amdgcn_tanhf)
  return __builtin_amdgcn_tanhf(v);          // v_tanh_f32
#elif __has_builtin(__builtin_amdgcn_tanh_f32)
  return __builtin_amdgcn_tanh_f32(v);
#else
  return tanhf(v);
#endif
}

// ---------------------------------------------------------------------------
// Prep: swizzle Wh (fp32 [512,512], k-major rows) into bf16 WMMA B-fragments.
//   lane L holds column N = L%16; half i holds K = (L<16 ? 0 : 16) + i.
// Flat index: ((w*16 + kc)*4 + nt)*512 + lane*16 + i   (w = wave = N/64)
// ---------------------------------------------------------------------------
__global__ __launch_bounds__(256) void rnn_prep_bfrag(const float* __restrict__ Wh,
                                                      __bf16* __restrict__ Bfrag) {
  int idx  = blockIdx.x * 256 + threadIdx.x;     // 262144 total
  int i    = idx & 15;
  int lane = (idx >> 4) & 31;
  int nt   = (idx >> 9) & 3;
  int kc   = (idx >> 11) & 15;
  int w    = idx >> 15;
  int n = w * 64 + nt * 16 + (lane & 15);
  int k = kc * 32 + ((lane < 16) ? 0 : 16) + i;
  Bfrag[idx] = (__bf16)Wh[k * HID + n];
}

// ---------------------------------------------------------------------------
// Main persistent-per-tile RNN kernel. 64 blocks x 256 threads (8 waves).
// ---------------------------------------------------------------------------
__global__ __launch_bounds__(256) void rnn_step_kernel(
    const float*  __restrict__ x,       // [B, S]
    const float*  __restrict__ Wx,      // [H]
    const float*  __restrict__ Wo,      // [H, 10]
    const float*  __restrict__ bh,      // [H]
    const float*  __restrict__ bo,      // [10]
    const float*  __restrict__ h_init,  // [B, H]
    const __bf16* __restrict__ Bfrag,   // [8][16][4][32][16] bf16
    float*        __restrict__ out)     // [B, 10]
{
  extern __shared__ char smem[];
  __bf16* hb  = (__bf16*)smem;                   // [2][BTILE][HSTRIDE]
  float*  xs  = (float*)(smem + XS_OFF);         // [BTILE]
  __bf16* Bsh = (__bf16*)(smem + BSH_OFF);       // [w][kc 0..7][nt][lane][16]

  const int tid  = threadIdx.x;
  const int wave = tid >> 5;
  const int lane = tid & 31;
  const int l16  = lane & 15;
  const bool hi  = lane >= 16;
  const int b0   = blockIdx.x * BTILE;

  // --- park the low-K half of the swizzled weights in LDS (one 1KB block/thread) ---
  {
    const int w  = tid >> 5;
    const int kc = (tid >> 2) & 7;     // kc 0..7
    const int nt = tid & 3;
    const uint4* src = (const uint4*)(Bfrag + (((w * 16 + kc) * 4 + nt) << 9));
    uint4*       dst = (uint4*)(Bsh + (((w * 8 + kc) * 4 + nt) << 9));
    for (int q = 0; q < 64; ++q) dst[q] = src[q];
  }

  // --- init h state from h_init (fp32 -> bf16) ---
  for (int e = tid; e < BTILE * HID; e += 256) {
    int m = e >> 9;                    // / 512
    int k = e & (HID - 1);
    hb[m * HSTRIDE + k] = (__bf16)h_init[(b0 + m) * HID + k];
  }

  // --- per-lane epilogue constants (this wave owns columns [wave*64, +64)) ---
  const int nbase = wave * 64;
  float wxv[4], bhv[4];
#pragma unroll
  for (int nt = 0; nt < 4; ++nt) {
    int n = nbase + nt * 16 + l16;
    wxv[nt] = Wx[n];
    bhv[nt] = bh[n];
  }
  const __bf16* Bw = Bfrag + (size_t)wave * (16 * 4 * 32 * 16);
  const int aoff = hi ? 16 : 0;

  int cur = 0;
  __syncthreads();

  for (int t = 0; t < SEQ; ++t) {
    if (tid < BTILE) xs[tid] = x[(b0 + tid) * SEQ + t];
    __syncthreads();

    // Launder offsets so loop-invariant weight loads are NOT hoisted
    // into VGPRs across the t-loop (prevents register spill). 32-bit
    // offsets keep address-space inference intact (ds_ vs global_).
    unsigned lws = wave << 14;         // wave base into Bsh (halves)
    unsigned gof = 0;                  // extra offset into Bfrag (halves)
    asm volatile("" : "+v"(lws), "+v"(gof));

    v8f Cacc[4] = {{}, {}, {}, {}};
    const __bf16* arow = hb + (cur * BTILE + l16) * HSTRIDE;
    const __bf16* Bls  = Bsh + lws;
    const __bf16* Bgs  = Bw + gof;

    // ---- K-chunks 0..7: B resident in LDS ----
#pragma unroll
    for (int kc = 0; kc < 8; ++kc) {
      union { v16bf v; v8bf h[2]; } A;
      A.h[0] = *(const v8bf*)(arow + kc * 32 + aoff);       // ds_load_b128
      A.h[1] = *(const v8bf*)(arow + kc * 32 + 8 + aoff);   // ds_load_b128
#pragma unroll
      for (int nt = 0; nt < 4; ++nt) {
        v16bf B = *(const v16bf*)(Bls + (((kc * 4 + nt) * 32 + lane) << 4));
        Cacc[nt] = __builtin_amdgcn_wmma_f32_16x16x32_bf16(
            false, A.v, false, B, (short)0, Cacc[nt], false, false);
      }
    }

    // ---- K-chunks 8..15: B streamed from L2-resident fragment buffer ----
#pragma unroll
    for (int kc = 8; kc < 16; ++kc) {
      union { v16bf v; v8bf h[2]; } A;
      A.h[0] = *(const v8bf*)(arow + kc * 32 + aoff);
      A.h[1] = *(const v8bf*)(arow + kc * 32 + 8 + aoff);
#pragma unroll
      for (int nt = 0; nt < 4; ++nt) {
        v16bf B = *(const v16bf*)(Bgs + (((kc * 4 + nt) * 32 + lane) << 4));
        Cacc[nt] = __builtin_amdgcn_wmma_f32_16x16x32_bf16(
            false, A.v, false, B, (short)0, Cacc[nt], false, false);
      }
    }

    // ---- epilogue: + x*Wx + bh, tanh (v_tanh_f32), bf16, write to other buffer ----
    // C layout: VGPR j, lanes 0-15 -> M=j, N=lane; lanes 16-31 -> M=j+8, N=lane-16
    const int nxt = cur ^ 1;
#pragma unroll
    for (int nt = 0; nt < 4; ++nt) {
      const int n = nbase + nt * 16 + l16;
#pragma unroll
      for (int j = 0; j < 8; ++j) {
        const int m = (hi ? 8 : 0) + j;
        float v = Cacc[nt][j] + xs[m] * wxv[nt] + bhv[nt];
        hb[(nxt * BTILE + m) * HSTRIDE + n] = (__bf16)fast_tanh(v);
      }
    }
    cur = nxt;
    __syncthreads();
  }

  // --- output GEMM: out[b, c] = h_final[b, :] @ Wo[:, c] + bo[c] ---
  if (tid < BTILE * NCLS) {
    const int m = tid / NCLS;
    const int c = tid % NCLS;
    float s = bo[c];
    for (int k = 0; k < HID; ++k)
      s += (float)hb[(cur * BTILE + m) * HSTRIDE + k] * Wo[k * NCLS + c];
    out[(b0 + m) * NCLS + c] = s;
  }
}

// ---------------------------------------------------------------------------
extern "C" void kernel_launch(void* const* d_in, const int* in_sizes, int n_in,
                              void* d_out, int out_size, void* d_ws, size_t ws_size,
                              hipStream_t stream) {
  const float* x      = (const float*)d_in[0];   // [1024,1024]
  const float* Wx     = (const float*)d_in[1];   // [1,512]
  const float* Wh     = (const float*)d_in[2];   // [512,512]
  const float* Wo     = (const float*)d_in[3];   // [512,10]
  const float* bh     = (const float*)d_in[4];   // [512]
  const float* bo     = (const float*)d_in[5];   // [10]
  const float* h_init = (const float*)d_in[6];   // [1024,512]
  float* out = (float*)d_out;                    // [1024,10]

  __bf16* Bfrag = (__bf16*)d_ws;                 // 512 KB of workspace

  rnn_prep_bfrag<<<262144 / 256, 256, 0, stream>>>(Wh, Bfrag);

  rnn_step_kernel<<<1024 / BTILE, 256, SMEM_BYTES, stream>>>(
      x, Wx, Wo, bh, bo, h_init, Bfrag, out);
}
